// FullGraphRel_39436389712080
// MI455X (gfx1250) — matrix operand order repaired and checked by reading, not compiled, not run on hardware
//
#include <hip/hip_runtime.h>

// ---------------------------------------------------------------------------
// Problem constants
// ---------------------------------------------------------------------------
#define Bsz   8
#define Rdim  40
#define Ldim  128
#define INs   256
#define DREL  15
#define Fnum  200
#define HIDs  300
#define KPAD  64             // energy K padded 40 -> 64
#define FPAD  224            // filters padded 200 -> 224 (multiple of 32)
#define KCONV 2304           // 256*9
#define OUTSP 15876          // 126*126
#define NCOLS 126

#define ASTR  72             // LDS row stride (u16), K-tile 64: 16B-aligned frag runs
#define BSTR  72             // LDS col stride (u16) for transposed B tile

// ---------------------------------------------------------------------------
// WMMA types / helpers (CDNA5 gfx1250, wave32)
// ---------------------------------------------------------------------------
typedef __attribute__((ext_vector_type(16))) __bf16 v16bf;
typedef __attribute__((ext_vector_type(8)))  float  v8f;

__device__ __forceinline__ unsigned short f2bf(float f) {
  unsigned u = __float_as_uint(f);
  unsigned r = u + 0x7FFFu + ((u >> 16) & 1u);   // round-to-nearest-even
  return (unsigned short)(r >> 16);
}
__device__ __forceinline__ float leakyf(float x) { return x > 0.f ? x : 0.01f * x; }
__device__ __forceinline__ unsigned fmapu(float f) {
  unsigned u = __float_as_uint(f);
  return (u >> 31) ? ~u : (u | 0x80000000u);     // monotone float->uint
}
__device__ __forceinline__ float funmap(unsigned u) {
  return __uint_as_float((u >> 31) ? (u & 0x7FFFFFFFu) : ~u);
}

// Async DMA: one 16B chunk global -> LDS per lane, tracked by ASYNCcnt.
__device__ __forceinline__ void async_b128_to_lds(const unsigned short* gsrc,
                                                  unsigned short* ldst) {
  const unsigned lo = (unsigned)(unsigned long long)ldst;   // LDS byte address
  asm volatile("global_load_async_to_lds_b128 %0, %1, off"
               :: "v"(lo), "v"(gsrc) : "memory");
}
__device__ __forceinline__ void wait_async0() {
  asm volatile("s_wait_asynccnt 0x0" ::: "memory");
}

// A tile in LDS: Asl[row][kk] (row = M, K-tile 64, stride ASTR).
// Lane frag at K-sub ksub: two contiguous 8x bf16 runs at kk=ksub+h8, +16.
__device__ __forceinline__ v16bf load_a_frag(const unsigned short* lds, int rowBase,
                                             int ksub, int lane) {
  union { v16bf v; uint4 q[2]; } a;
  const unsigned short* p =
      lds + (rowBase + (lane & 15)) * ASTR + ksub + ((lane >> 4) << 3);
  a.q[0] = *(const uint4*)(p);        // ds_load_b128
  a.q[1] = *(const uint4*)(p + 16);   // ds_load_b128
  return a.v;
}
// B tile in LDS TRANSPOSED: Bsl[col][k], stride BSTR; frag = two 8x bf16 runs at k=ksub+h8, +16.
__device__ __forceinline__ v16bf load_b_frag(const unsigned short* lds, int colBase,
                                             int ksub, int lane) {
  union { v16bf v; uint4 q[2]; } b;
  const unsigned short* p =
      lds + (colBase + (lane & 15)) * BSTR + ksub + ((lane >> 4) << 3);
  b.q[0] = *(const uint4*)(p);        // ds_load_b128
  b.q[1] = *(const uint4*)(p + 16);   // ds_load_b128
  return b.v;
}
#define WMMA_BF16(A, B, C) \
  __builtin_amdgcn_wmma_f32_16x16x32_bf16(false, (A), false, (B), (short)0, (C), false, false)

// ---------------------------------------------------------------------------
// Workspace layout (bytes, 256-aligned regions)
// ---------------------------------------------------------------------------
#define OFF_MARG  ((size_t)0)                    // [131072] f32
#define OFF_ET    ((size_t)524288)               // [131072][64] bf16
#define OFF_WHBF  ((size_t)17301504)             // [1024][512] bf16
#define OFF_WPQ   ((size_t)18350080)             // [512][512]  bf16
#define OFF_MMAT  ((size_t)18874368)             // [64][256]   bf16
#define OFF_PQ    ((size_t)18907136)             // [1024][512] f32 (P | Q)
#define OFF_H     ((size_t)21004288)             // [8][300] f32
#define OFF_TR    ((size_t)21014016)             // [8][200] f32
#define OFF_RESU  ((size_t)21020672)             // [8][200] u32
#define OFF_FILT  ((size_t)21027328)             // [8][224][2304] bf16
#define OFF_MEMB  ((size_t)29284864)             // [8][256][128][128] bf16

// ---------------------------------------------------------------------------
// 1) relu(energy): marginal + bf16 transposed/padded energy rows (16B stores)
// ---------------------------------------------------------------------------
__global__ __launch_bounds__(256) void k_prep(const float* __restrict__ energy,
                                              float* __restrict__ marg,
                                              unsigned short* __restrict__ eT) {
  const int idx = blockIdx.x * 256 + threadIdx.x;            // (b,i,j) flat
  const int b = idx >> 14, ij = idx & 16383;
  const float* ep = energy + (size_t)b * Rdim * 16384 + ij;
  unsigned short* er = eT + (size_t)idx * KPAD;
  float s = 0.f;
#pragma unroll
  for (int g = 0; g < 5; ++g) {                              // 5*8 = 40 relations
    union { uint4 q; unsigned short u[8]; } t;
#pragma unroll
    for (int r = 0; r < 8; ++r) {
      float e = ep[(size_t)(g * 8 + r) * 16384];
      e = e > 0.f ? e : 0.f;
      s += e;
      t.u[r] = f2bf(e);
    }
    *(uint4*)(er + g * 8) = t.q;
  }
  const uint4 z = {0u, 0u, 0u, 0u};
#pragma unroll
  for (int g = 5; g < 8; ++g) *(uint4*)(er + g * 8) = z;     // pad 40 -> 64
  marg[idx] = s;
}

// ---------------------------------------------------------------------------
// 2) small converts / transposes
// ---------------------------------------------------------------------------
__global__ __launch_bounds__(256) void k_cvt_wh(const float* __restrict__ wh,
                                                unsigned short* __restrict__ whbf) {
  const int i8 = (blockIdx.x * 256 + threadIdx.x) * 8;
  const float4 a = *(const float4*)(wh + i8);
  const float4 b = *(const float4*)(wh + i8 + 4);
  union { uint4 q; unsigned short u[8]; } t;
  t.u[0] = f2bf(a.x); t.u[1] = f2bf(a.y); t.u[2] = f2bf(a.z); t.u[3] = f2bf(a.w);
  t.u[4] = f2bf(b.x); t.u[5] = f2bf(b.y); t.u[6] = f2bf(b.z); t.u[7] = f2bf(b.w);
  *(uint4*)(whbf + i8) = t.q;
}

__global__ __launch_bounds__(256) void k_wpq(const float* __restrict__ arcW,
                                             unsigned short* __restrict__ wpq) {
  const int idx = blockIdx.x * 256 + threadIdx.x;            // 512*512/8
  const int k = idx >> 6, nb = (idx & 63) * 8;
  union { uint4 q; unsigned short u[8]; } t;
#pragma unroll
  for (int u = 0; u < 8; ++u) {
    const int n = nb + u;
    const float v = (n < 256) ? arcW[n * 1039 + k]                 // P-half
                              : arcW[(n - 256) * 1039 + 512 + k];  // Q-half
    t.u[u] = f2bf(v);
  }
  *(uint4*)(wpq + k * 512 + nb) = t.q;
}

__global__ __launch_bounds__(256) void k_mmat(const float* __restrict__ relE,
                                              const float* __restrict__ arcW,
                                              unsigned short* __restrict__ mmat) {
  const int idx = blockIdx.x * 256 + threadIdx.x;            // 64*256
  const int r = idx >> 8, o = idx & 255;
  float s = 0.f;
  if (r < Rdim) {
#pragma unroll
    for (int d = 0; d < DREL; ++d) s += relE[r * DREL + d] * arcW[o * 1039 + 1024 + d];
  }
  mmat[r * 256 + o] = f2bf(s);
}

// zero filter padding rows + max accumulators (must rerun every launch)
__global__ __launch_bounds__(256) void k_zero(unsigned short* __restrict__ filt,
                                              unsigned* __restrict__ resU) {
  const int idx = blockIdx.x * 256 + threadIdx.x;            // 55296 vec8 elems
  if (idx < Bsz * Fnum) resU[idx] = 0u;
  const int b = idx / 6912, rr = idx - b * 6912;
  const uint4 z = {0u, 0u, 0u, 0u};
  *(uint4*)(filt + (size_t)b * (FPAD * KCONV) + (size_t)Fnum * KCONV + rr * 8) = z;
}

// ---------------------------------------------------------------------------
// 3) tiny MLPs
// ---------------------------------------------------------------------------
__global__ __launch_bounds__(512) void k_mlp(const float* __restrict__ e1,
                                             const float* __restrict__ e2,
                                             const float* __restrict__ ffW,
                                             const float* __restrict__ ffb,
                                             const float* __restrict__ itW,
                                             const float* __restrict__ itb,
                                             float* __restrict__ h,
                                             float* __restrict__ tr) {
  __shared__ float ins[1024];
  const int b = blockIdx.x, tid = threadIdx.x;
  ins[tid] = e1[b * 512 + tid];
  ins[tid + 512] = e2[b * 512 + tid];
  __syncthreads();
  if (tid < HIDs) {
    float a = ffb[tid];
    const float* w = ffW + (size_t)tid * 1024;
    for (int k = 0; k < 1024; ++k) a += ins[k] * w[k];
    h[b * HIDs + tid] = leakyf(a);
  } else if (tid < HIDs + Fnum) {
    const int f = tid - HIDs;
    float a = itb[f];
    const float* w = itW + (size_t)f * 1024;
    for (int k = 0; k < 1024; ++k) a += ins[k] * w[k];
    tr[b * Fnum + f] = leakyf(a);
  }
}

// ---------------------------------------------------------------------------
// 4) WMMA GEMM: PQ[1024,512] = word_h_bf16[1024,512] x wpq[512,512]
//    256 thr = 8 waves as 2(M) x 4(N); tile M32 x N64, stage K=64 (2 WMMA/stage)
// ---------------------------------------------------------------------------
__global__ __launch_bounds__(256) void k_gemm_pq(const unsigned short* __restrict__ A,
                                                 const unsigned short* __restrict__ B,
                                                 float* __restrict__ C) {
  __shared__ __align__(16) unsigned short Asl[32 * ASTR];
  __shared__ __align__(16) unsigned short Bsl[64 * BSTR];
  const int tid = threadIdx.x, lane = tid & 31, wave = tid >> 5;
  const int mw = wave >> 2, nw = wave & 3;
  const int m0 = blockIdx.y * 32, n0 = blockIdx.x * 64;
  const int arow = tid >> 3, akk = (tid & 7) << 3;           // A: 16B chunk per lane
  const int bcol = tid & 63, bkb = (tid >> 6) << 4;          // B: col, 16 k's
  v8f acc = {};
  for (int k0 = 0; k0 < 512; k0 += 64) {
    async_b128_to_lds(A + (m0 + arow) * 512 + k0 + akk, Asl + arow * ASTR + akk);
    {
      union { uint4 q; unsigned short u[8]; } t0, t1;
#pragma unroll
      for (int u = 0; u < 8; ++u) t0.u[u] = B[(k0 + bkb + u) * 512 + n0 + bcol];
#pragma unroll
      for (int u = 0; u < 8; ++u) t1.u[u] = B[(k0 + bkb + 8 + u) * 512 + n0 + bcol];
      *(uint4*)(Bsl + bcol * BSTR + bkb) = t0.q;
      *(uint4*)(Bsl + bcol * BSTR + bkb + 8) = t1.q;
    }
    wait_async0();
    __syncthreads();
    acc = WMMA_BF16(load_a_frag(Asl, mw * 16, 0, lane),
                    load_b_frag(Bsl, nw * 16, 0, lane), acc);
    acc = WMMA_BF16(load_a_frag(Asl, mw * 16, 32, lane),
                    load_b_frag(Bsl, nw * 16, 32, lane), acc);
    __syncthreads();
  }
  const int col = n0 + nw * 16 + (lane & 15);
  const int rb = m0 + mw * 16 + ((lane >> 4) << 3);
#pragma unroll
  for (int v = 0; v < 8; ++v) C[(rb + v) * 512 + col] = acc[v];
}

// ---------------------------------------------------------------------------
// 5) WMMA GEMM (relation term, K=64 single stage) + fused arc epilogue
//    memb[b,o,i,j] = bf16(leaky(marg*(P[b,j,o]+Q[b,i,o]) + acc + arc_b[o]))
// ---------------------------------------------------------------------------
__global__ __launch_bounds__(256) void k_gemm_arc(const unsigned short* __restrict__ A,
                                                  const unsigned short* __restrict__ B,
                                                  const float* __restrict__ marg,
                                                  const float* __restrict__ PQ,
                                                  const float* __restrict__ bias,
                                                  unsigned short* __restrict__ memb) {
  __shared__ __align__(16) unsigned short Asl[32 * ASTR];
  __shared__ __align__(16) unsigned short Bsl[64 * BSTR];
  const int tid = threadIdx.x, lane = tid & 31, wave = tid >> 5;
  const int mw = wave >> 2, nw = wave & 3;
  const int m0 = blockIdx.y * 32, n0 = blockIdx.x * 64;
  const int arow = tid >> 3, akk = (tid & 7) << 3;
  const int bcol = tid & 63, bkb = (tid >> 6) << 4;
  async_b128_to_lds(A + (size_t)(m0 + arow) * KPAD + akk, Asl + arow * ASTR + akk);
  {
    union { uint4 q; unsigned short u[8]; } t0, t1;
#pragma unroll
    for (int u = 0; u < 8; ++u) t0.u[u] = B[(bkb + u) * 256 + n0 + bcol];
#pragma unroll
    for (int u = 0; u < 8; ++u) t1.u[u] = B[(bkb + 8 + u) * 256 + n0 + bcol];
    *(uint4*)(Bsl + bcol * BSTR + bkb) = t0.q;
    *(uint4*)(Bsl + bcol * BSTR + bkb + 8) = t1.q;
  }
  wait_async0();
  __syncthreads();
  v8f acc = {};
  acc = WMMA_BF16(load_a_frag(Asl, mw * 16, 0, lane),
                  load_b_frag(Bsl, nw * 16, 0, lane), acc);
  acc = WMMA_BF16(load_a_frag(Asl, mw * 16, 32, lane),
                  load_b_frag(Bsl, nw * 16, 32, lane), acc);
  const int o  = n0 + nw * 16 + (lane & 15);
  const int rb = m0 + mw * 16 + ((lane >> 4) << 3);
  const float bo = bias[o];
#pragma unroll
  for (int v = 0; v < 8; ++v) {
    const int row = rb + v;
    const int b = row >> 14, rem = row & 16383, i = rem >> 7, j = rem & 127;
    const float m = marg[row];
    const float p = PQ[(b * 128 + j) * 512 + o];
    const float q = PQ[(b * 128 + i) * 512 + 256 + o];
    const float val = leakyf(m * (p + q) + acc[v] + bo);
    memb[((size_t)(b * 256 + o) * 128 + i) * 128 + j] = f2bf(val);
  }
}

// ---------------------------------------------------------------------------
// 6) filter factory: one wave per fw1 row (coalesced), shfl-sum, bf16 out
// ---------------------------------------------------------------------------
__global__ __launch_bounds__(256) void k_genfilt(const float* __restrict__ fw1,
                                                 const float* __restrict__ fb1,
                                                 const float* __restrict__ h,
                                                 unsigned short* __restrict__ filt) {
  __shared__ float hs[Bsz * HIDs];
  const int tid = threadIdx.x;
  for (int i = tid; i < Bsz * HIDs; i += 256) hs[i] = h[i];
  __syncthreads();
  const int wave = tid >> 5, lane = tid & 31;
  const long row = (long)blockIdx.x * 8 + wave;              // 460800 rows total
  if (row >= (long)Fnum * INs * 9) return;
  const float* wr = fw1 + row * HIDs;
  float acc[Bsz] = {};
  for (int k = lane; k < HIDs; k += 32) {
    const float w = wr[k];
#pragma unroll
    for (int b = 0; b < Bsz; ++b) acc[b] += w * hs[b * HIDs + k];
  }
#pragma unroll
  for (int b = 0; b < Bsz; ++b)
#pragma unroll
    for (int m = 1; m < 32; m <<= 1) acc[b] += __shfl_xor(acc[b], m, 32);
  if (lane == 0) {
    const float bias = fb1[row];
#pragma unroll
    for (int b = 0; b < Bsz; ++b)
      filt[(long)b * (FPAD * KCONV) + row] = f2bf(leakyf(acc[b] + bias));
  }
}

// ---------------------------------------------------------------------------
// 7) implicit-GEMM conv + fused leaky + spatial max (dominant phase)
//    per block: b, 32 filters (2 m-waves) x 64 spatial (4 n-waves),
//    K=2304 in stages of 64 (2 WMMA per barrier pair, async A DMA)
// ---------------------------------------------------------------------------
__global__ __launch_bounds__(256) void k_conv(const unsigned short* __restrict__ memb,
                                              const unsigned short* __restrict__ filt,
                                              unsigned* __restrict__ resU) {
  __shared__ __align__(16) unsigned short Asl[32 * ASTR];
  __shared__ __align__(16) unsigned short Bsl[64 * BSTR];
  const int b = blockIdx.z;
  const int m0 = blockIdx.y * 32;
  const int n0 = blockIdx.x * 64;
  const int tid = threadIdx.x, lane = tid & 31, wave = tid >> 5;
  const int mw = wave >> 2, nw = wave & 3;
  const unsigned short* fb = filt + (size_t)b * (FPAD * KCONV);
  const unsigned short* xb = memb + (size_t)b * (INs * Ldim * Ldim);
  const int arow = tid >> 3, akk = (tid & 7) << 3;
  // B loader: each thread owns one spatial column, 16 consecutive K values
  const int nl = tid & 63, kb = (tid >> 6) << 4;
  const int n = n0 + nl;
  const bool ok = n < OUTSP;
  const int y = ok ? (n / NCOLS) : 0;
  const int x = ok ? (n - y * NCOLS) : 0;
  const unsigned short* xp = xb + y * Ldim + x;              // base of this column
  v8f acc = {};
  for (int k0 = 0; k0 < KCONV; k0 += 64) {
    // A: filter rows, 16B chunk per lane via async DMA (global -> LDS direct)
    async_b128_to_lds(fb + (size_t)(m0 + arow) * KCONV + k0 + akk,
                      Asl + arow * ASTR + akk);
    { // B: im2col gather -> transposed LDS (per-k coalesced across lanes)
      union { uint4 q; unsigned short u[8]; } t0, t1;
#pragma unroll
      for (int u = 0; u < 8; ++u) {
        const int k = k0 + kb + u;
        const int c = k / 9, rr = k - c * 9, dy = rr / 3, dx = rr - dy * 3;
        t0.u[u] = ok ? xp[(size_t)c * (Ldim * Ldim) + dy * Ldim + dx] : 0;
      }
#pragma unroll
      for (int u = 0; u < 8; ++u) {
        const int k = k0 + kb + 8 + u;
        const int c = k / 9, rr = k - c * 9, dy = rr / 3, dx = rr - dy * 3;
        t1.u[u] = ok ? xp[(size_t)c * (Ldim * Ldim) + dy * Ldim + dx] : 0;
      }
      *(uint4*)(Bsl + nl * BSTR + kb) = t0.q;
      *(uint4*)(Bsl + nl * BSTR + kb + 8) = t1.q;
    }
    wait_async0();
    __syncthreads();
    acc = WMMA_BF16(load_a_frag(Asl, mw * 16, 0, lane),
                    load_b_frag(Bsl, nw * 16, 0, lane), acc);
    acc = WMMA_BF16(load_a_frag(Asl, mw * 16, 32, lane),
                    load_b_frag(Bsl, nw * 16, 32, lane), acc);
    __syncthreads();
  }
  // epilogue: leaky, mask invalid cols, max across 16 cols per row, atomic max
  const int ncol = n0 + nw * 16 + (lane & 15);
  const bool colok = ncol < OUTSP;
  const int fbase = m0 + mw * 16 + ((lane >> 4) << 3);
#pragma unroll
  for (int v = 0; v < 8; ++v) {
    float val = colok ? leakyf(acc[v]) : -3.4e38f;
#pragma unroll
    for (int m = 1; m < 16; m <<= 1) val = fmaxf(val, __shfl_xor(val, m, 32));
    if ((lane & 15) == 0) {
      const int f = fbase + v;
      if (f < Fnum) atomicMax(&resU[b * Fnum + f], fmapu(val));
    }
  }
}

// ---------------------------------------------------------------------------
// 8) final: out = transformed + decoded max
// ---------------------------------------------------------------------------
__global__ __launch_bounds__(256) void k_final(const float* __restrict__ tr,
                                               const unsigned* __restrict__ resU,
                                               float* __restrict__ out) {
  const int i = blockIdx.x * 256 + threadIdx.x;
  if (i < Bsz * Fnum) out[i] = tr[i] + funmap(resU[i]);
}

// ---------------------------------------------------------------------------
extern "C" void kernel_launch(void* const* d_in, const int* in_sizes, int n_in,
                              void* d_out, int out_size, void* d_ws, size_t ws_size,
                              hipStream_t stream) {
  const float* energy = (const float*)d_in[0];
  const float* word_h = (const float*)d_in[1];
  const float* e1     = (const float*)d_in[2];
  const float* e2     = (const float*)d_in[3];
  // d_in[4] = sent_len (unused by forward path)
  const float* relE   = (const float*)d_in[5];
  const float* arcW   = (const float*)d_in[6];
  const float* arcb   = (const float*)d_in[7];
  const float* ffW    = (const float*)d_in[8];
  const float* ffb    = (const float*)d_in[9];
  const float* fw1    = (const float*)d_in[10];
  const float* fb1    = (const float*)d_in[11];
  const float* itW    = (const float*)d_in[12];
  const float* itb    = (const float*)d_in[13];
  float* out = (float*)d_out;

  char* ws = (char*)d_ws;
  float*          marg = (float*)(ws + OFF_MARG);
  unsigned short* eT   = (unsigned short*)(ws + OFF_ET);
  unsigned short* whbf = (unsigned short*)(ws + OFF_WHBF);
  unsigned short* wpq  = (unsigned short*)(ws + OFF_WPQ);
  unsigned short* mmat = (unsigned short*)(ws + OFF_MMAT);
  float*          PQ   = (float*)(ws + OFF_PQ);
  float*          h    = (float*)(ws + OFF_H);
  float*          tr   = (float*)(ws + OFF_TR);
  unsigned*       resU = (unsigned*)(ws + OFF_RESU);
  unsigned short* filt = (unsigned short*)(ws + OFF_FILT);
  unsigned short* memb = (unsigned short*)(ws + OFF_MEMB);

  // phase 0: prep + converts + zeroing (independent)
  k_prep  <<<512, 256, 0, stream>>>(energy, marg, eT);
  k_cvt_wh<<<256, 256, 0, stream>>>(word_h, whbf);
  k_wpq   <<<128, 256, 0, stream>>>(arcW, wpq);
  k_mmat  <<<64,  256, 0, stream>>>(relE, arcW, mmat);
  k_zero  <<<216, 256, 0, stream>>>(filt, resU);
  k_mlp   <<<Bsz, 512, 0, stream>>>(e1, e2, ffW, ffb, itW, itb, h, tr);

  // phase 1: P|Q GEMM (bf16 WMMA)
  k_gemm_pq<<<dim3(512 / 64, 1024 / 32), 256, 0, stream>>>(whbf, wpq, PQ);

  // phase 2: relation GEMM + fused arc epilogue -> mem_bank (bf16)
  k_gemm_arc<<<dim3(256 / 64, 131072 / 32), 256, 0, stream>>>(eT, mmat, marg, PQ, arcb, memb);

  // phase 3: filter factory (bandwidth bound on fw1, 553 MB)
  k_genfilt<<<57600, 256, 0, stream>>>(fw1, fb1, h, filt);

  // phase 4: implicit-GEMM conv + fused leaky/max-pool (dominant: 117 GFLOP WMMA)
  k_conv<<<dim3((OUTSP + 63) / 64, FPAD / 32, Bsz), 256, 0, stream>>>(memb, filt, resU);

  // phase 5: combine
  k_final<<<(Bsz * Fnum + 255) / 256, 256, 0, stream>>>(tr, resU, out);
}
